// MinRNN_44624710205571
// MI455X (gfx1250) — matrix operand order repaired and checked by reading, not compiled, not run on hardware
//
#include <hip/hip_runtime.h>
#include <hip/hip_bf16.h>

// ---------------------------------------------------------------------------
// MinRNN on MI455X (gfx1250):
//   K0: convert sentence f32 -> f16 (64 MiB), pack Wf/Wi/Wh f32 -> f16 (1.5 MiB)
//   K1: fused 3-projection GEMM with v_wmma_f32_16x16x32_f16, epilogue computes
//       f' = sig(f)/(sig(f)+sig(i)),  s = sig(i)*h~/(sig(f)+sig(i))
//       stored in scan-friendly [t][b][u] layout (f32).
//   K2: segmented linear scan (8 segments x 256 steps) -> affine (P,S) pairs
//   K3: combine 8 affine pairs -> h_final[b][u]
//   K4: head 512->64->1 + sigmoid
// ---------------------------------------------------------------------------

typedef __attribute__((ext_vector_type(16))) _Float16 v16h;
typedef __attribute__((ext_vector_type(8)))  _Float16 v8h;
typedef __attribute__((ext_vector_type(4)))  _Float16 v4h;
typedef __attribute__((ext_vector_type(8)))  float    v8f;

#define BB 32
#define TT 2048
#define EE 512
#define UU 512
#define ROWS (BB*TT)          // 65536

// ------------------------- workspace layout (bytes) ------------------------
#define X16_OFF  ((size_t)0)                         // 65536*512*2  = 64 MiB
#define W16_OFF  (X16_OFF + (size_t)ROWS*EE*2)       // 3*512*512*2  = 1.5 MiB
#define FP_OFF   (W16_OFF + (size_t)3*UU*EE*2)       // 65536*512*4  = 128 MiB
#define INP_OFF  (FP_OFF  + (size_t)ROWS*UU*4)       // 128 MiB
#define P_OFF    (INP_OFF + (size_t)ROWS*UU*4)       // 8*16384*4 = 512 KiB
#define S_OFF    (P_OFF   + (size_t)8*BB*UU*4)
#define H_OFF    (S_OFF   + (size_t)8*BB*UU*4)       // 32*512*4 = 64 KiB

// --------------------------- K0a: sentence -> f16 ---------------------------
__global__ void cvt_x_kernel(const float* __restrict__ x, _Float16* __restrict__ y, int n4) {
    int i = blockIdx.x * blockDim.x + threadIdx.x;
    if (i < n4) {
        float4 v = ((const float4*)x)[i];
        v4h h;
        h[0] = (_Float16)v.x; h[1] = (_Float16)v.y;
        h[2] = (_Float16)v.z; h[3] = (_Float16)v.w;
        ((v4h*)y)[i] = h;
    }
}

// --------------------------- K0b: weights -> f16 ----------------------------
__global__ void cvt_w_kernel(const float* __restrict__ Wf, const float* __restrict__ Wi,
                             const float* __restrict__ Wh, _Float16* __restrict__ y) {
    int i = blockIdx.x * blockDim.x + threadIdx.x;   // 0 .. 3*512*512-1
    int g = i >> 18;                                 // 262144 elements per matrix
    int r = i & 262143;
    const float* src = (g == 0) ? Wf : (g == 1) ? Wi : Wh;
    y[i] = (_Float16)src[r];
}

// ----------------- K1: fused 3-way WMMA GEMM + gate epilogue ----------------
// grid = (4096, 4), block = 256 (8 waves). Each wave owns one 16x16 tile of
// all three projections (shared A fragment -> 3 WMMAs per K-chunk).
__global__ __launch_bounds__(256) void gates_gemm_kernel(
    const _Float16* __restrict__ x16,   // [65536][512]
    const _Float16* __restrict__ w16,   // [3][512][512]  (Wf, Wi, Wh rows, row-major)
    const float* __restrict__ bf, const float* __restrict__ bi, const float* __restrict__ bh,
    float* __restrict__ fpb, float* __restrict__ inpb)
{
    const int wave  = threadIdx.x >> 5;
    const int lane  = threadIdx.x & 31;
    const int strip = blockIdx.x;                         // 16-row strip, 0..4095
    const int ntile = ((blockIdx.y << 3) | wave) << 4;    // 16-col tile base, 0..496
    const int lr    = lane & 15;
    const int hh    = lane >> 4;

    // A fragment: lane holds row (strip*16 + lr); 16 contiguous K starting at hh*16
    const _Float16* ap  = x16 + (size_t)((strip << 4) + lr) * EE;
    const int aoff = hh * 16;

    // B fragment: lane holds W row (ntile + lr); 16 contiguous K starting at hh*16
    const _Float16* wf = w16 + (size_t)(ntile + lr) * EE;
    const _Float16* wi = wf + (size_t)UU * EE;
    const _Float16* wh = wi + (size_t)UU * EE;
    const int boff = hh * 16;

    v8f accF = {}; v8f accI = {}; v8f accH = {};
    #pragma unroll 4
    for (int kb = 0; kb < EE; kb += 32) {
        v8h a0 = *(const v8h*)(ap + kb + aoff);
        v8h a1 = *(const v8h*)(ap + kb + 8 + aoff);
        v16h a = __builtin_shufflevector(a0, a1, 0,1,2,3,4,5,6,7,8,9,10,11,12,13,14,15);
        v16h bF = *(const v16h*)(wf + kb + boff);
        v16h bI = *(const v16h*)(wi + kb + boff);
        v16h bH = *(const v16h*)(wh + kb + boff);
        accF = __builtin_amdgcn_wmma_f32_16x16x32_f16(false, a, false, bF, (short)0, accF, false, false);
        accI = __builtin_amdgcn_wmma_f32_16x16x32_f16(false, a, false, bI, (short)0, accI, false, false);
        accH = __builtin_amdgcn_wmma_f32_16x16x32_f16(false, a, false, bH, (short)0, accH, false, false);
    }

    // Epilogue: C/D layout -> lane owns column n = ntile+lr, rows j + hh*8.
    const int   n   = ntile + lr;
    const float bfv = bf[n], biv = bi[n], bhv = bh[n];
    const int   r0  = strip << 4;
    const int   bix = r0 >> 11;        // batch index (constant per strip)
    const int   t0  = r0 & (TT - 1);

    #pragma unroll
    for (int j = 0; j < 8; ++j) {
        int   m  = j + hh * 8;
        float fv = 1.0f / (1.0f + __expf(-(accF[j] + bfv)));
        float iv = 1.0f / (1.0f + __expf(-(accI[j] + biv)));
        float hv = accH[j] + bhv;
        float dn = fv + iv;
        size_t o = ((size_t)(t0 + m) * BB + bix) * UU + n;   // [t][b][u]
        fpb[o]  = fv / dn;
        inpb[o] = (iv * hv) / dn;
    }
}

// ------------------ K2: segmented scan -> affine (P, S) ---------------------
// h_out = P * h_in + S over 256 steps; 8 segments * 16384 (b,u) lanes.
__global__ __launch_bounds__(256) void seg_scan_kernel(
    const float* __restrict__ fpb, const float* __restrict__ inpb,
    float* __restrict__ Pb, float* __restrict__ Sb)
{
    int tid = blockIdx.x * 256 + threadIdx.x;   // 0 .. 131071
    int idx = tid & 16383;                      // b*512 + u
    int s   = tid >> 14;                        // segment
    const float* f = fpb  + (size_t)s * 256 * 16384 + idx;
    const float* g = inpb + (size_t)s * 256 * 16384 + idx;
    float P = 1.0f, S = 0.0f;
    #pragma unroll 8
    for (int t = 0; t < 256; ++t) {
        float fv = f[(size_t)t * 16384];
        float gv = g[(size_t)t * 16384];
        S = fmaf(fv, S, gv);
        P *= fv;
    }
    Pb[tid] = P;
    Sb[tid] = S;
}

// ------------------ K3: combine 8 affine segments ---------------------------
__global__ void combine_kernel(const float* __restrict__ Pb, const float* __restrict__ Sb,
                               float* __restrict__ hfin)
{
    int idx = blockIdx.x * 256 + threadIdx.x;   // 0 .. 16383
    float h = 0.0f;
    #pragma unroll
    for (int s = 0; s < 8; ++s)
        h = fmaf(Pb[s * 16384 + idx], h, Sb[s * 16384 + idx]);
    hfin[idx] = h;
}

// ------------------ K4: head  512 -> 64 -> 1 + sigmoid ----------------------
__global__ __launch_bounds__(64) void head_kernel(
    const float* __restrict__ hfin, const float* __restrict__ W1, const float* __restrict__ b1,
    const float* __restrict__ W2, const float* __restrict__ b2, float* __restrict__ out)
{
    __shared__ float red[64];
    int b = blockIdx.x, j = threadIdx.x;
    const float* h = hfin + (size_t)b * UU;
    const float* w = W1 + (size_t)j * UU;
    float acc = b1[j];
    #pragma unroll 4
    for (int e = 0; e < UU; ++e) acc = fmaf(h[e], w[e], acc);
    red[j] = acc * W2[j];
    __syncthreads();
    for (int off = 32; off > 0; off >>= 1) {
        if (j < off) red[j] += red[j + off];
        __syncthreads();
    }
    if (j == 0) out[b] = 1.0f / (1.0f + __expf(-(red[0] + b2[0])));
}

// ---------------------------------------------------------------------------
extern "C" void kernel_launch(void* const* d_in, const int* in_sizes, int n_in,
                              void* d_out, int out_size, void* d_ws, size_t ws_size,
                              hipStream_t stream) {
    (void)in_sizes; (void)n_in; (void)out_size; (void)ws_size;
    const float* sentence = (const float*)d_in[0];
    const float* Wf = (const float*)d_in[1];
    const float* bf = (const float*)d_in[2];
    const float* Wi = (const float*)d_in[3];
    const float* bi = (const float*)d_in[4];
    const float* Wh = (const float*)d_in[5];
    const float* bh = (const float*)d_in[6];
    const float* W1 = (const float*)d_in[7];
    const float* b1 = (const float*)d_in[8];
    const float* W2 = (const float*)d_in[9];
    const float* b2 = (const float*)d_in[10];
    float* out = (float*)d_out;

    char* ws = (char*)d_ws;
    _Float16* x16  = (_Float16*)(ws + X16_OFF);
    _Float16* w16  = (_Float16*)(ws + W16_OFF);
    float*    fpb  = (float*)(ws + FP_OFF);
    float*    inpb = (float*)(ws + INP_OFF);
    float*    Pb   = (float*)(ws + P_OFF);
    float*    Sb   = (float*)(ws + S_OFF);
    float*    hfin = (float*)(ws + H_OFF);

    // K0: conversions
    {
        int n4 = ROWS * EE / 4;                       // 8388608 float4s
        cvt_x_kernel<<<(n4 + 255) / 256, 256, 0, stream>>>(sentence, x16, n4);
        cvt_w_kernel<<<(3 * UU * EE) / 256, 256, 0, stream>>>(Wf, Wi, Wh, w16);
    }
    // K1: fused gate GEMMs (WMMA)
    {
        dim3 grid(ROWS / 16, 4);
        gates_gemm_kernel<<<grid, 256, 0, stream>>>(x16, w16, bf, bi, bh, fpb, inpb);
    }
    // K2: segmented scan
    seg_scan_kernel<<<(8 * BB * UU) / 256, 256, 0, stream>>>(fpb, inpb, Pb, Sb);
    // K3: combine
    combine_kernel<<<(BB * UU) / 256, 256, 0, stream>>>(Pb, Sb, hfin);
    // K4: head
    head_kernel<<<BB, 64, 0, stream>>>(hfin, W1, b1, W2, b2, out);
}
